// StructureModule_13675175870803
// MI455X (gfx1250) — compile-verified, matrix operand order checked by reading, not run on hardware
//
#include <hip/hip_runtime.h>
#include <hip/hip_bf16.h>
#include <math.h>
#include <stdint.h>

typedef __attribute__((ext_vector_type(16))) __bf16 v16bf;
typedef __attribute__((ext_vector_type(8)))  float  v8f;

#define LSEQ 1024
#define CDIM 384
#define PDIM 128
#define HH   12
#define CATW 2112   // 192 r_s + 288 f_p + 96 f_n + 1536 r_2d
#define NWAVE 4
#define BT    (NWAVE*32)

// ---- workspace layout (float elements) ----
#define OFF_QS   0
#define N_QS     (HH*LSEQ*16)
#define OFF_KS   (OFF_QS + N_QS)
#define OFF_VS   (OFF_KS + N_QS)
#define OFF_QP   (OFF_VS + N_QS)
#define N_QP     (LSEQ*HH*12)
#define OFF_KP   (OFF_QP + N_QP)
#define OFF_VP   (OFF_KP + N_QP)
#define N_VP     (LSEQ*HH*24)
#define OFF_SQQ  (OFF_VP + N_VP)
#define OFF_SQK  (OFF_SQQ + HH*LSEQ)
#define OFF_WPB  (OFF_SQK + HH*LSEQ)     // 2048 bf16 = 1024 float slots
#define OFF_CAT  (OFF_WPB + 1024)        // L*2112 concat features

#define SCALAR_W 0.14433757f   // 1/sqrt(48)
#define POINT_WC 0.13608277f   // 1/sqrt(54)
#define PAIR_W   0.57735027f   // 1/sqrt(3)

// ---- gfx1250 async global->LDS path (probe-confirmed: builtin exists, takes
// address-space-qualified int4-vector pointers) ----
typedef int v4i __attribute__((vector_size(16)));
typedef __attribute__((address_space(1))) v4i as1_v4i;
typedef __attribute__((address_space(3))) v4i as3_v4i;
#if defined(__gfx1250__) && __has_builtin(__builtin_amdgcn_global_load_async_to_lds_b128)
#define HAVE_ASYNC 1
__device__ __forceinline__ void async_ld_b128(const float* g, float* l) {
  __builtin_amdgcn_global_load_async_to_lds_b128(
      (as1_v4i*)(uintptr_t)g, (as3_v4i*)(unsigned int)(uintptr_t)l, 0, 0);
}
__device__ __forceinline__ void async_wait0() {
#if __has_builtin(__builtin_amdgcn_s_wait_asynccnt)
  __builtin_amdgcn_s_wait_asynccnt(0);
#else
  asm volatile("s_wait_asynccnt 0x0" ::: "memory");
#endif
}
#endif

// ---------------- kernel 0: pack pair_w*Wpair into WMMA-B bf16 fragments ----
__global__ __launch_bounds__(256) void prep_wpair(const float* __restrict__ Wpair,
                                                  float* __restrict__ ws) {
  __bf16* wb = (__bf16*)(ws + OFF_WPB);
  for (int e = threadIdx.x; e < 4 * 32 * 16; e += 256) {
    int kk = e >> 9, rem = e & 511, lane = rem >> 4, s = rem & 15;
    int kl = s + (s >= 8 ? 8 : 0) + (lane >= 16 ? 8 : 0);  // K within 32-block
    int K = kk * 32 + kl;
    int N = lane & 15;
    float v = (N < HH) ? PAIR_W * Wpair[K * HH + N] : 0.0f;
    wb[e] = (__bf16)v;
  }
}

// ---------------- kernel 1: projections + rigid + norms ---------------------
__global__ __launch_bounds__(256) void proj_kernel(
    const float* __restrict__ in1d, const float* __restrict__ rot,
    const float* __restrict__ trans,
    const float* __restrict__ Wq,  const float* __restrict__ bq,
    const float* __restrict__ Wkv, const float* __restrict__ bkv,
    const float* __restrict__ Wqp, const float* __restrict__ bqp,
    const float* __restrict__ Wkvp,const float* __restrict__ bkvp,
    float* __restrict__ ws) {
  int i = blockIdx.x, t = threadIdx.x;
  __shared__ float xs[CDIM];
  __shared__ float rawq[144];
  __shared__ float rawkv[432];
  __shared__ float sqp[48];
  __shared__ float sqkp[48];
  for (int k = t; k < CDIM; k += 256) xs[k] = in1d[(size_t)i * CDIM + k];
  __syncthreads();

  for (int col = t; col < 1152; col += 256) {
    if (col < 192) {
      float acc = bq[col];
      for (int k = 0; k < CDIM; k++) acc += xs[k] * Wq[(size_t)k * 192 + col];
      int h = col >> 4, c = col & 15;
      ws[OFF_QS + ((size_t)h * LSEQ + i) * 16 + c] = acc * SCALAR_W;
    } else if (col < 576) {
      int c = col - 192;
      float acc = bkv[c];
      for (int k = 0; k < CDIM; k++) acc += xs[k] * Wkv[(size_t)k * 384 + c];
      int h = c >> 5, m = c & 31;
      if (m < 16) ws[OFF_KS + ((size_t)h * LSEQ + i) * 16 + m] = acc;
      else        ws[OFF_VS + ((size_t)h * LSEQ + i) * 16 + (m - 16)] = acc;
    } else if (col < 720) {
      int c = col - 576;
      float acc = bqp[c];
      for (int k = 0; k < CDIM; k++) acc += xs[k] * Wqp[(size_t)k * 144 + c];
      rawq[c] = acc;
    } else {
      int c = col - 720;
      float acc = bkvp[c];
      for (int k = 0; k < CDIM; k++) acc += xs[k] * Wkvp[(size_t)k * 432 + c];
      rawkv[c] = acc;
    }
  }
  __syncthreads();

  const float* R = rot + (size_t)i * 9;
  float t0 = trans[i * 3 + 0], t1 = trans[i * 3 + 1], t2 = trans[i * 3 + 2];
  if (t < 48) {  // query points
    float p0 = rawq[t], p1 = rawq[48 + t], p2 = rawq[96 + t];
    float gx = R[0]*p0 + R[1]*p1 + R[2]*p2 + t0;
    float gy = R[3]*p0 + R[4]*p1 + R[5]*p2 + t1;
    float gz = R[6]*p0 + R[7]*p1 + R[8]*p2 + t2;
    int h = t >> 2, n = t & 3;
    float* q = ws + OFF_QP + ((size_t)i * HH + h) * 12 + n * 3;
    q[0] = gx; q[1] = gy; q[2] = gz;
    sqp[t] = gx*gx + gy*gy + gz*gz;
  }
  if (t < 144) {  // key + value points
    float p0 = rawkv[t], p1 = rawkv[144 + t], p2 = rawkv[288 + t];
    float gx = R[0]*p0 + R[1]*p1 + R[2]*p2 + t0;
    float gy = R[3]*p0 + R[4]*p1 + R[5]*p2 + t1;
    float gz = R[6]*p0 + R[7]*p1 + R[8]*p2 + t2;
    int h = t / 12, m = t % 12;
    if (m < 4) {
      float* kpp = ws + OFF_KP + ((size_t)i * HH + h) * 12 + m * 3;
      kpp[0] = gx; kpp[1] = gy; kpp[2] = gz;
      sqkp[h * 4 + m] = gx*gx + gy*gy + gz*gz;
    } else {
      float* vpp = ws + OFF_VP + ((size_t)i * HH + h) * 24 + (m - 4) * 3;
      vpp[0] = gx; vpp[1] = gy; vpp[2] = gz;
    }
  }
  __syncthreads();
  if (t < HH) {
    ws[OFF_SQQ + (size_t)t * LSEQ + i] = sqp[4*t] + sqp[4*t+1] + sqp[4*t+2] + sqp[4*t+3];
    ws[OFF_SQK + (size_t)t * LSEQ + i] = sqkp[4*t] + sqkp[4*t+1] + sqkp[4*t+2] + sqkp[4*t+3];
  }
}

// ---------------- kernel 2: fused flash-IPA attention -----------------------
// Per block: one query i. 4 waves, each owning a 256-wide j slice.
// Per 32-j tile: stage inputs_2d tile to LDS (async TDM-path copies), WMMA
// pair logits on top of a VALU scalar+point base, online softmax, WMMA r_2d
// accumulation (probabilities feed A straight from registers), VALU r_s/r_p.
__global__ __launch_bounds__(BT) void fused_attn(
    const float* __restrict__ in2d, const unsigned char* __restrict__ mask,
    const float* __restrict__ tpw, const float* __restrict__ bpair,
    const float* __restrict__ rot, const float* __restrict__ trans,
    float* __restrict__ ws) {
  int i = blockIdx.x, t = threadIdx.x;
  int wave = t >> 5, lane = t & 31;
  int h = lane & 15, hi = lane >> 4;
  bool hv = h < HH;

  __shared__ float tile[NWAVE * 32 * PDIM];   // per-wave 32 x 128 fp32 tile
  __shared__ float ptile[NWAVE * 32 * 16];    // per-wave prob tile [jj][h]
  __shared__ float wcorr[NWAVE * 16];
  __shared__ float statm[NWAVE * 16];
  __shared__ float statl[NWAVE * 16];
  __shared__ float wscale[NWAVE * 16];
  __shared__ float r2dacc[HH * PDIM];
  __shared__ float sacc[480];

  const float* qs  = ws + OFF_QS;
  const float* ks  = ws + OFF_KS;
  const float* vs  = ws + OFF_VS;
  const float* qp  = ws + OFF_QP;
  const float* kp  = ws + OFF_KP;
  const float* vp  = ws + OFF_VP;
  const float* sqq = ws + OFF_SQQ;
  const float* sqk = ws + OFF_SQK;
  const v16bf* wb  = (const v16bf*)(ws + OFF_WPB);

  // per-lane query-side data
  float qrow[16], qpt[12];
  float sq_i = 0.f, pwh = 0.f, pb = 0.f;
  if (hv) {
    const float* qr = qs + ((size_t)h * LSEQ + i) * 16;
    #pragma unroll
    for (int c = 0; c < 16; c++) qrow[c] = qr[c];
    const float* qq = qp + ((size_t)i * HH + h) * 12;
    #pragma unroll
    for (int r = 0; r < 12; r++) qpt[r] = qq[r];
    sq_i = sqq[(size_t)h * LSEQ + i];
    float tw = tpw[h];
    float sp = tw > 20.f ? tw : log1pf(__expf(tw));
    pwh = -0.5f * POINT_WC * sp;
    pb  = PAIR_W * bpair[h];
  }

  // scalar accumulator slot metadata (480 slots over BT threads, 4 each)
  int s_u[4], s_h[4];
  #pragma unroll
  for (int q = 0; q < 4; q++) {
    int u = t + BT * q;
    s_u[q] = u;
    s_h[q] = (u < 192) ? (u >> 4) : (u < 480 ? (u - 192) / 24 : 0);
  }

  float m_run = -3.0e38f, l_run = 0.f;
  v8f creg[8] = {};
  float racc[4] = {0.f, 0.f, 0.f, 0.f};

  float* mytile = tile + wave * (32 * PDIM);
  float* myp    = ptile + wave * (32 * 16);

  for (int jb = 0; jb < 8; jb++) {
    int j0 = wave * 256 + jb * 32;

    // ---- stage inputs_2d[i, j0:j0+32, :] into LDS ----
#ifdef HAVE_ASYNC
    for (int r = 0; r < 32; r++) {
      const float* g = in2d + ((size_t)i * LSEQ + j0 + r) * PDIM + lane * 4;
      async_ld_b128(g, mytile + r * PDIM + lane * 4);
    }
    async_wait0();
#else
    for (int r = 0; r < 32; r++) {
      const float4 v4 = *(const float4*)(in2d + ((size_t)i * LSEQ + j0 + r) * PDIM + lane * 4);
      *(float4*)(mytile + r * PDIM + lane * 4) = v4;
    }
#endif

    // ---- logits for two 16-j subtiles: VALU base + WMMA pair bias ----
    v8f carr[2];
    #pragma unroll
    for (int st = 0; st < 2; st++) {
      v8f cc = {};
      #pragma unroll
      for (int v = 0; v < 8; v++) {
        float base = 0.f;
        if (hv) {
          int j = j0 + st * 16 + v + 8 * hi;
          if (mask[j]) {
            const float* kr = ks + ((size_t)h * LSEQ + j) * 16;
            float d = 0.f;
            #pragma unroll
            for (int c2 = 0; c2 < 16; c2++) d += qrow[c2] * kr[c2];
            const float* kpr = kp + ((size_t)j * HH + h) * 12;
            float pq = 0.f;
            #pragma unroll
            for (int r2 = 0; r2 < 12; r2++) pq += qpt[r2] * kpr[r2];
            base = d + pwh * (sq_i + sqk[(size_t)h * LSEQ + j] - 2.f * pq) + pb;
          } else {
            base = -3.0e38f;
          }
        }
        cc[v] = base;
      }
      int row = st * 16 + h;             // A row = local j
      const float* ar = mytile + row * PDIM;
      #pragma unroll
      for (int kk = 0; kk < 4; kk++) {
        v16bf a;
        int b0 = kk * 32 + (hi << 3);
        #pragma unroll
        for (int s = 0; s < 8; s++) a[s] = (__bf16)ar[b0 + s];
        #pragma unroll
        for (int s = 0; s < 8; s++) a[8 + s] = (__bf16)ar[b0 + 16 + s];
        cc = __builtin_amdgcn_wmma_f32_16x16x32_bf16(false, a, false,
                                                     wb[kk * 32 + lane],
                                                     (short)0, cc, false, false);
      }
      carr[st] = cc;
    }

    // ---- online softmax stats (per head, both lane-halves kept coherent) --
    float tmax = -3.0e38f;
    #pragma unroll
    for (int v = 0; v < 8; v++) tmax = fmaxf(tmax, fmaxf(carr[0][v], carr[1][v]));
    tmax = fmaxf(tmax, __shfl_xor(tmax, 16, 32));
    float newm = fmaxf(m_run, tmax);
    float corr = __expf(m_run - newm);
    m_run = newm;

    float p0[8], p1[8], lsum = 0.f;
    #pragma unroll
    for (int v = 0; v < 8; v++) {
      p0[v] = __expf(carr[0][v] - m_run);
      p1[v] = __expf(carr[1][v] - m_run);
      lsum += p0[v] + p1[v];
    }
    lsum += __shfl_xor(lsum, 16, 32);
    l_run = l_run * corr + lsum;

    if (hi == 0) wcorr[wave * 16 + h] = corr;
    #pragma unroll
    for (int v = 0; v < 8; v++) {
      myp[(v + 8 * hi) * 16 + h]      = p0[v];
      myp[(16 + v + 8 * hi) * 16 + h] = p1[v];
    }
    __syncthreads();

    // ---- rescale running accumulators ----
    #pragma unroll
    for (int v = 0; v < 8; v++) {
      float cv = wcorr[wave * 16 + v + 8 * hi];
      #pragma unroll
      for (int pt = 0; pt < 8; pt++) creg[pt][v] *= cv;
    }
    #pragma unroll
    for (int q = 0; q < 4; q++)
      if (s_u[q] < 480) racc[q] *= wcorr[wave * 16 + s_h[q]];

    // ---- r_2d WMMA: probabilities feed A directly from registers ----
    v16bf a;
    #pragma unroll
    for (int s = 0; s < 8; s++) { a[s] = (__bf16)p0[s]; a[8 + s] = (__bf16)p1[s]; }
    #pragma unroll
    for (int pt = 0; pt < 8; pt++) {
      v16bf b;
      #pragma unroll
      for (int s = 0; s < 16; s++) {
        int kl = s + (s >= 8 ? 8 : 0) + (lane >= 16 ? 8 : 0);
        b[s] = (__bf16)mytile[kl * PDIM + pt * 16 + h];
      }
      creg[pt] = __builtin_amdgcn_wmma_f32_16x16x32_bf16(false, a, false, b,
                                                         (short)0, creg[pt],
                                                         false, false);
    }

    // ---- r_s / r_p VALU accumulation ----
    for (int jj = 0; jj < 32; jj++) {
      int j = j0 + jj;
      #pragma unroll
      for (int q = 0; q < 4; q++) {
        int u = s_u[q];
        if (u < 192) {
          racc[q] += myp[jj * 16 + s_h[q]] *
                     vs[((size_t)s_h[q] * LSEQ + j) * 16 + (u & 15)];
        } else if (u < 480) {
          int rem = (u - 192) % 24;
          racc[q] += myp[jj * 16 + s_h[q]] *
                     vp[((size_t)j * HH + s_h[q]) * 24 + rem];
        }
      }
    }
    __syncthreads();
  }

  // ---- merge the 4 per-wave softmax partials (deterministic) ----
  if (hi == 0) { statm[wave * 16 + h] = m_run; statl[wave * 16 + h] = l_run; }
  for (int idx = t; idx < HH * PDIM; idx += BT) r2dacc[idx] = 0.f;
  for (int idx = t; idx < 480; idx += BT) sacc[idx] = 0.f;
  __syncthreads();
  if (t < NWAVE * 16) {
    int h2 = t & 15;
    float mg = -3.0e38f;
    for (int w3 = 0; w3 < NWAVE; w3++) mg = fmaxf(mg, statm[w3 * 16 + h2]);
    float lg = 0.f;
    for (int w3 = 0; w3 < NWAVE; w3++)
      lg += __expf(statm[w3 * 16 + h2] - mg) * statl[w3 * 16 + h2];
    wscale[t] = __expf(statm[t] - mg) / lg;
  }
  __syncthreads();
  for (int w2 = 0; w2 < NWAVE; w2++) {
    if (wave == w2) {
      #pragma unroll
      for (int v = 0; v < 8; v++) {
        int hr = v + 8 * hi;
        if (hr < HH) {
          float scv = wscale[wave * 16 + hr];
          #pragma unroll
          for (int pt = 0; pt < 8; pt++)
            r2dacc[hr * PDIM + pt * 16 + h] += scv * creg[pt][v];
        }
      }
      #pragma unroll
      for (int q = 0; q < 4; q++)
        if (s_u[q] < 480) sacc[s_u[q]] += wscale[wave * 16 + s_h[q]] * racc[q];
    }
    __syncthreads();
  }

  // ---- emit concat features ----
  float* cat = ws + OFF_CAT + (size_t)i * CATW;
  for (int u = t; u < 192; u += BT) cat[u] = sacc[u];
  for (int idx = t; idx < HH * PDIM; idx += BT) cat[576 + idx] = r2dacc[idx];
  if (t < 96) {  // inverse rigid + norms on r_p
    int hh = t >> 3, n = t & 7;
    const float* R = rot + (size_t)i * 9;
    float d0 = sacc[192 + hh * 24 + n * 3 + 0] - trans[i * 3 + 0];
    float d1 = sacc[192 + hh * 24 + n * 3 + 1] - trans[i * 3 + 1];
    float d2 = sacc[192 + hh * 24 + n * 3 + 2] - trans[i * 3 + 2];
    float lx = R[0]*d0 + R[3]*d1 + R[6]*d2;
    float ly = R[1]*d0 + R[4]*d1 + R[7]*d2;
    float lz = R[2]*d0 + R[5]*d1 + R[8]*d2;
    cat[192 + 0 * 96 + hh * 8 + n] = lx;
    cat[192 + 1 * 96 + hh * 8 + n] = ly;
    cat[192 + 2 * 96 + hh * 8 + n] = lz;
    cat[480 + hh * 8 + n] = sqrtf(lx*lx + ly*ly + lz*lz + 1e-8f);
  }
}

// ---------------- kernel 3: final projection --------------------------------
__global__ __launch_bounds__(256) void final_kernel(
    const float* __restrict__ Wout, const float* __restrict__ bout,
    const float* __restrict__ ws, float* __restrict__ out) {
  int i = blockIdx.x, t = threadIdx.x;
  __shared__ float row[CATW];
  const float* cat = ws + OFF_CAT + (size_t)i * CATW;
  for (int k = t; k < CATW; k += 256) row[k] = cat[k];
  __syncthreads();
  for (int c = t; c < CDIM; c += 256) {
    float acc = bout[c];
    for (int k = 0; k < CATW; k++) acc += row[k] * Wout[(size_t)k * CDIM + c];
    out[(size_t)i * CDIM + c] = acc;
  }
}

extern "C" void kernel_launch(void* const* d_in, const int* in_sizes, int n_in,
                              void* d_out, int out_size, void* d_ws, size_t ws_size,
                              hipStream_t stream) {
  const float* in1d  = (const float*)d_in[0];
  const float* in2d  = (const float*)d_in[1];
  const float* rot   = (const float*)d_in[2];
  const float* trans = (const float*)d_in[3];
  const unsigned char* mask = (const unsigned char*)d_in[4];
  const float* Wq    = (const float*)d_in[5];
  const float* bq    = (const float*)d_in[6];
  const float* Wkv   = (const float*)d_in[7];
  const float* bkv   = (const float*)d_in[8];
  const float* Wqp   = (const float*)d_in[9];
  const float* bqp   = (const float*)d_in[10];
  const float* Wkvp  = (const float*)d_in[11];
  const float* bkvp  = (const float*)d_in[12];
  const float* Wpair = (const float*)d_in[13];
  const float* bpair = (const float*)d_in[14];
  const float* tpw   = (const float*)d_in[15];
  const float* Wout  = (const float*)d_in[16];
  const float* bout  = (const float*)d_in[17];
  float* out = (float*)d_out;
  float* ws  = (float*)d_ws;

  prep_wpair<<<1, 256, 0, stream>>>(Wpair, ws);
  proj_kernel<<<LSEQ, 256, 0, stream>>>(in1d, rot, trans, Wq, bq, Wkv, bkv,
                                        Wqp, bqp, Wkvp, bkvp, ws);
  fused_attn<<<LSEQ, BT, 0, stream>>>(in2d, mask, tpw, bpair, rot, trans, ws);
  final_kernel<<<LSEQ, 256, 0, stream>>>(Wout, bout, ws, out);
}
